// UpdateSpeaker_61778809585745
// MI455X (gfx1250) — matrix operand order repaired and checked by reading, not compiled, not run on hardware
//
#include <hip/hip_runtime.h>
#include <hip/hip_bf16.h>
#include <stdint.h>

#define HID 1024
#define UD  512
#define NB  16384

typedef __attribute__((ext_vector_type(16))) __bf16 v16bf;
typedef __attribute__((ext_vector_type(8)))  float  v8f;

union FragB16 {
  uint32_t u[8];
  v16bf    v;
};

// Native f32 -> bf16 (compiler emits v_cvt_pk_bf16_f32)
__device__ __forceinline__ uint32_t pack_bf(float a, float b) {
  union { __bf16 h[2]; uint32_t u; } r;
  r.h[0] = (__bf16)a;
  r.h[1] = (__bf16)b;
  return r.u;
}

__device__ __forceinline__ uint16_t f2bf(float f) {
  union { __bf16 h; uint16_t u; } r;
  r.h = (__bf16)f;
  return r.u;
}

// A-matrix (16x32 bf16) K index for vgpr v, lane-half h (ISA 7.12.2 table)
__device__ __forceinline__ int kbA(int v, int h) {
  return (v < 4) ? (h * 8 + 2 * v) : (16 + h * 8 + 2 * (v - 4));
}
// B-matrix (32x16 bf16) K index for vgpr v, lane-half h
__device__ __forceinline__ int kbB(int v, int h) {
  return (v < 4) ? (h * 8 + 2 * v) : (16 + h * 8 + 2 * (v - 4));
}

__device__ __forceinline__ float sigmoidf_fast(float x) {
  return 1.0f / (1.0f + __expf(-x));
}

// ---------------------------------------------------------------------------
// fp32 -> bf16 conversion (4 elements / thread, vector loads)
// ---------------------------------------------------------------------------
__global__ void cvt_f32_bf16_kernel(const float4* __restrict__ src,
                                    uint2* __restrict__ dst, int n4) {
  int i = blockIdx.x * blockDim.x + threadIdx.x;
  if (i < n4) {
    float4 f = src[i];
    uint2 o;
    o.x = pack_bf(f.x, f.y);
    o.y = pack_bf(f.z, f.w);
    dst[i] = o;
  }
}

// ---------------------------------------------------------------------------
// Kernel 1: pre-activations of r, p, z gates + hx = enc @ WA_xh^T
//   acc_r = enc@Wxr^T + spk@Whr^T + adr@Vhr^T   (same for p, z)
//   stores: rs = sigmoid(r)*spk (bf16), pa = sigmoid(p)*adr (bf16),
//           z = sigmoid(z) (f32), hx (f32)
// Block: 256 thr (8 waves, 4x2 over MxN), tile 128x64, per-wave 32x32.
// All GEMM operands are pre-converted bf16: fragment loads are pure
// packed-dword loads (merge to global_load_b128), no VALU in the hot loop.
// ---------------------------------------------------------------------------
__global__ __launch_bounds__(256) void gates_kernel(
    const uint16_t* __restrict__ encB,  // [NB x HID] bf16
    const uint16_t* __restrict__ spkB,  // [NB x UD]  bf16
    const uint16_t* __restrict__ adrB,  // [NB x UD]  bf16
    const float* __restrict__ spk, const float* __restrict__ adr,  // f32 (epilogue)
    const uint16_t* __restrict__ Wx,   // 4 x [UD x HID] bf16, order r,p,z,h
    const uint16_t* __restrict__ Wh,   // 3 x [UD x UD]  bf16, order r,p,z
    const uint16_t* __restrict__ Vh,   // 3 x [UD x UD]  bf16, order r,p,z
    uint16_t* __restrict__ rs, uint16_t* __restrict__ pa,
    float* __restrict__ zbuf, float* __restrict__ hx)
{
  const int lane = threadIdx.x & 31;
  const int wave = threadIdx.x >> 5;
  const int wm = wave & 3;
  const int wn = wave >> 2;
  const int n_block = blockIdx.x & 7;    // 8 blocks of 64 cols -> 512
  const int m_block = blockIdx.x >> 3;   // 128 blocks of 128 rows -> 16384
  const int mBase = m_block * 128 + wm * 32;
  const int nBase = n_block * 64 + wn * 32;

  const int lm = lane & 15;
  const int h  = lane >> 4;

  const v8f vzero = {0.f, 0.f, 0.f, 0.f, 0.f, 0.f, 0.f, 0.f};
  v8f acc[4][2][2];
#pragma unroll
  for (int o = 0; o < 4; ++o)
#pragma unroll
    for (int mi = 0; mi < 2; ++mi)
#pragma unroll
      for (int ni = 0; ni < 2; ++ni) acc[o][mi][ni] = vzero;

  // ---- enc [NB x HID] against 4 weight matrices (r,p,z,h share A frags)
  for (int k0 = 0; k0 < HID; k0 += 32) {
    FragB16 a[2];
#pragma unroll
    for (int mi = 0; mi < 2; ++mi) {
      const uint16_t* arow = encB + (size_t)(mBase + mi * 16 + lm) * HID + k0;
#pragma unroll
      for (int v = 0; v < 8; ++v)
        a[mi].u[v] = *(const uint32_t*)(arow + kbA(v, h));
    }
#pragma unroll
    for (int o = 0; o < 4; ++o) {
      const uint16_t* W = Wx + (size_t)o * UD * HID;
#pragma unroll
      for (int ni = 0; ni < 2; ++ni) {
        FragB16 b;
        const uint16_t* wrow = W + (size_t)(nBase + ni * 16 + lm) * HID + k0;
#pragma unroll
        for (int v = 0; v < 8; ++v)
          b.u[v] = *(const uint32_t*)(wrow + kbB(v, h));
#pragma unroll
        for (int mi = 0; mi < 2; ++mi)
          acc[o][mi][ni] = __builtin_amdgcn_wmma_f32_16x16x32_bf16(
              false, a[mi].v, false, b.v, (short)0, acc[o][mi][ni], false, false);
      }
    }
  }

  // ---- spk [NB x UD] against Wh{r,p,z}
  for (int k0 = 0; k0 < UD; k0 += 32) {
    FragB16 a[2];
#pragma unroll
    for (int mi = 0; mi < 2; ++mi) {
      const uint16_t* arow = spkB + (size_t)(mBase + mi * 16 + lm) * UD + k0;
#pragma unroll
      for (int v = 0; v < 8; ++v)
        a[mi].u[v] = *(const uint32_t*)(arow + kbA(v, h));
    }
#pragma unroll
    for (int o = 0; o < 3; ++o) {
      const uint16_t* W = Wh + (size_t)o * UD * UD;
#pragma unroll
      for (int ni = 0; ni < 2; ++ni) {
        FragB16 b;
        const uint16_t* wrow = W + (size_t)(nBase + ni * 16 + lm) * UD + k0;
#pragma unroll
        for (int v = 0; v < 8; ++v)
          b.u[v] = *(const uint32_t*)(wrow + kbB(v, h));
#pragma unroll
        for (int mi = 0; mi < 2; ++mi)
          acc[o][mi][ni] = __builtin_amdgcn_wmma_f32_16x16x32_bf16(
              false, a[mi].v, false, b.v, (short)0, acc[o][mi][ni], false, false);
      }
    }
  }

  // ---- adr [NB x UD] against Vh{r,p,z}
  for (int k0 = 0; k0 < UD; k0 += 32) {
    FragB16 a[2];
#pragma unroll
    for (int mi = 0; mi < 2; ++mi) {
      const uint16_t* arow = adrB + (size_t)(mBase + mi * 16 + lm) * UD + k0;
#pragma unroll
      for (int v = 0; v < 8; ++v)
        a[mi].u[v] = *(const uint32_t*)(arow + kbA(v, h));
    }
#pragma unroll
    for (int o = 0; o < 3; ++o) {
      const uint16_t* W = Vh + (size_t)o * UD * UD;
#pragma unroll
      for (int ni = 0; ni < 2; ++ni) {
        FragB16 b;
        const uint16_t* wrow = W + (size_t)(nBase + ni * 16 + lm) * UD + k0;
#pragma unroll
        for (int v = 0; v < 8; ++v)
          b.u[v] = *(const uint32_t*)(wrow + kbB(v, h));
#pragma unroll
        for (int mi = 0; mi < 2; ++mi)
          acc[o][mi][ni] = __builtin_amdgcn_wmma_f32_16x16x32_bf16(
              false, a[mi].v, false, b.v, (short)0, acc[o][mi][ni], false, false);
      }
    }
  }

  // ---- epilogue: gates + gated activations
#pragma unroll
  for (int mi = 0; mi < 2; ++mi) {
#pragma unroll
    for (int ni = 0; ni < 2; ++ni) {
#pragma unroll
      for (int e = 0; e < 8; ++e) {
        const int row = mBase + mi * 16 + h * 8 + e;
        const int col = nBase + ni * 16 + lm;
        const size_t idx = (size_t)row * UD + col;
        const float rv = sigmoidf_fast(acc[0][mi][ni][e]);
        const float pv = sigmoidf_fast(acc[1][mi][ni][e]);
        const float zv = sigmoidf_fast(acc[2][mi][ni][e]);
        rs[idx]   = f2bf(rv * spk[idx]);
        pa[idx]   = f2bf(pv * adr[idx]);
        zbuf[idx] = zv;
        hx[idx]   = acc[3][mi][ni][e];
      }
    }
  }
}

// ---------------------------------------------------------------------------
// Kernel 2: h = hx + rs@Whh^T + pa@Vhh^T ; out = (1-z)*spk + z*tanh(h)
// ---------------------------------------------------------------------------
__global__ __launch_bounds__(256) void output_kernel(
    const uint16_t* __restrict__ rs, const uint16_t* __restrict__ pa,
    const uint16_t* __restrict__ Whh, const uint16_t* __restrict__ Vhh,
    const float* __restrict__ hx, const float* __restrict__ zbuf,
    const float* __restrict__ spk, float* __restrict__ out)
{
  const int lane = threadIdx.x & 31;
  const int wave = threadIdx.x >> 5;
  const int wm = wave & 3;
  const int wn = wave >> 2;
  const int n_block = blockIdx.x & 7;
  const int m_block = blockIdx.x >> 3;
  const int mBase = m_block * 128 + wm * 32;
  const int nBase = n_block * 64 + wn * 32;

  const int lm = lane & 15;
  const int h  = lane >> 4;

  const v8f vzero = {0.f, 0.f, 0.f, 0.f, 0.f, 0.f, 0.f, 0.f};
  v8f acc[2][2];
#pragma unroll
  for (int mi = 0; mi < 2; ++mi)
#pragma unroll
    for (int ni = 0; ni < 2; ++ni) acc[mi][ni] = vzero;

  for (int k0 = 0; k0 < UD; k0 += 32) {
    FragB16 a1[2], a2[2];
#pragma unroll
    for (int mi = 0; mi < 2; ++mi) {
      const uint16_t* r1 = rs + (size_t)(mBase + mi * 16 + lm) * UD + k0;
      const uint16_t* r2 = pa + (size_t)(mBase + mi * 16 + lm) * UD + k0;
#pragma unroll
      for (int v = 0; v < 8; ++v) {
        const int k = kbA(v, h);
        a1[mi].u[v] = *(const uint32_t*)(r1 + k);
        a2[mi].u[v] = *(const uint32_t*)(r2 + k);
      }
    }
#pragma unroll
    for (int ni = 0; ni < 2; ++ni) {
      FragB16 bw, bv;
      const uint16_t* wrow = Whh + (size_t)(nBase + ni * 16 + lm) * UD + k0;
      const uint16_t* vrow = Vhh + (size_t)(nBase + ni * 16 + lm) * UD + k0;
#pragma unroll
      for (int v = 0; v < 8; ++v) {
        bw.u[v] = *(const uint32_t*)(wrow + kbB(v, h));
        bv.u[v] = *(const uint32_t*)(vrow + kbB(v, h));
      }
#pragma unroll
      for (int mi = 0; mi < 2; ++mi) {
        acc[mi][ni] = __builtin_amdgcn_wmma_f32_16x16x32_bf16(
            false, a1[mi].v, false, bw.v, (short)0, acc[mi][ni], false, false);
        acc[mi][ni] = __builtin_amdgcn_wmma_f32_16x16x32_bf16(
            false, a2[mi].v, false, bv.v, (short)0, acc[mi][ni], false, false);
      }
    }
  }

#pragma unroll
  for (int mi = 0; mi < 2; ++mi) {
#pragma unroll
    for (int ni = 0; ni < 2; ++ni) {
#pragma unroll
      for (int e = 0; e < 8; ++e) {
        const int row = mBase + mi * 16 + h * 8 + e;
        const int col = nBase + ni * 16 + lm;
        const size_t idx = (size_t)row * UD + col;
        const float hv = acc[mi][ni][e] + hx[idx];
        const float zv = zbuf[idx];
        const float sv = spk[idx];
        out[idx] = (1.0f - zv) * sv + zv * tanhf(hv);
      }
    }
  }
}

// ---------------------------------------------------------------------------
extern "C" void kernel_launch(void* const* d_in, const int* in_sizes, int n_in,
                              void* d_out, int out_size, void* d_ws, size_t ws_size,
                              hipStream_t stream) {
  (void)in_sizes; (void)n_in; (void)out_size; (void)ws_size;

  const float* enc = (const float*)d_in[0];
  const float* spk = (const float*)d_in[1];
  const float* adr = (const float*)d_in[2];
  // d_in[3..6]  = WA_xr, WA_xp, WA_xz, WA_xh   [UD x HID]
  // d_in[7..10] = WA_hr, WA_hp, WA_hz, WA_hh   [UD x UD]
  // d_in[11..14]= VA_hr, VA_hp, VA_hz, VA_hh   [UD x UD]

  // Workspace layout (~176 MB):
  uint16_t* wsWx   = (uint16_t*)d_ws;                      // 4*UD*HID bf16
  uint16_t* wsWh   = wsWx   + (size_t)4 * UD * HID;        // 3*UD*UD  bf16 (r,p,z)
  uint16_t* wsVh   = wsWh   + (size_t)3 * UD * UD;         // 3*UD*UD  bf16 (r,p,z)
  uint16_t* wsWhh  = wsVh   + (size_t)3 * UD * UD;         // UD*UD bf16
  uint16_t* wsVhh  = wsWhh  + (size_t)UD * UD;             // UD*UD bf16
  uint16_t* wsEnc  = wsVhh  + (size_t)UD * UD;             // NB*HID bf16
  uint16_t* wsSpk  = wsEnc  + (size_t)NB * HID;            // NB*UD bf16
  uint16_t* wsAdr  = wsSpk  + (size_t)NB * UD;             // NB*UD bf16
  uint16_t* wsRs   = wsAdr  + (size_t)NB * UD;             // NB*UD bf16
  uint16_t* wsPa   = wsRs   + (size_t)NB * UD;             // NB*UD bf16
  float*    wsZ    = (float*)(wsPa + (size_t)NB * UD);     // NB*UD f32
  float*    wsHx   = wsZ + (size_t)NB * UD;                // NB*UD f32

  const int TC = 256;
  const int nWx4  = UD * HID / 4;   // 131072
  const int nWh4  = UD * UD / 4;    // 65536
  const int nEnc4 = NB * HID / 4;   // 4194304
  const int nUD4  = NB * UD / 4;    // 2097152

  // Weights
  for (int i = 0; i < 4; ++i)
    cvt_f32_bf16_kernel<<<(nWx4 + TC - 1) / TC, TC, 0, stream>>>(
        (const float4*)d_in[3 + i], (uint2*)(wsWx + (size_t)i * UD * HID), nWx4);
  for (int i = 0; i < 3; ++i)
    cvt_f32_bf16_kernel<<<(nWh4 + TC - 1) / TC, TC, 0, stream>>>(
        (const float4*)d_in[7 + i], (uint2*)(wsWh + (size_t)i * UD * UD), nWh4);
  for (int i = 0; i < 3; ++i)
    cvt_f32_bf16_kernel<<<(nWh4 + TC - 1) / TC, TC, 0, stream>>>(
        (const float4*)d_in[11 + i], (uint2*)(wsVh + (size_t)i * UD * UD), nWh4);
  cvt_f32_bf16_kernel<<<(nWh4 + TC - 1) / TC, TC, 0, stream>>>(
      (const float4*)d_in[10], (uint2*)wsWhh, nWh4);
  cvt_f32_bf16_kernel<<<(nWh4 + TC - 1) / TC, TC, 0, stream>>>(
      (const float4*)d_in[14], (uint2*)wsVhh, nWh4);
  // Activations
  cvt_f32_bf16_kernel<<<(nEnc4 + TC - 1) / TC, TC, 0, stream>>>(
      (const float4*)enc, (uint2*)wsEnc, nEnc4);
  cvt_f32_bf16_kernel<<<(nUD4 + TC - 1) / TC, TC, 0, stream>>>(
      (const float4*)spk, (uint2*)wsSpk, nUD4);
  cvt_f32_bf16_kernel<<<(nUD4 + TC - 1) / TC, TC, 0, stream>>>(
      (const float4*)adr, (uint2*)wsAdr, nUD4);

  // 128 M-blocks (128 rows each) x 8 N-blocks (64 cols each) = 1024 blocks
  const dim3 grid(1024), block(256);
  gates_kernel<<<grid, block, 0, stream>>>(wsEnc, wsSpk, wsAdr, spk, adr,
                                           wsWx, wsWh, wsVh,
                                           wsRs, wsPa, wsZ, wsHx);
  output_kernel<<<grid, block, 0, stream>>>(wsRs, wsPa, wsWhh, wsVhh, wsHx,
                                            wsZ, spk, (float*)d_out);
}
